// ProcessorLayer_17901423689759
// MI455X (gfx1250) — matrix operand order repaired and checked by reading, not compiled, not run on hardware
//
#include <hip/hip_runtime.h>
#include <hip/hip_bf16.h>
#include <math.h>

typedef __attribute__((ext_vector_type(16))) __bf16 v16bf;
typedef __attribute__((ext_vector_type(2)))  __bf16 v2bf;
typedef __attribute__((ext_vector_type(2)))  float  v2f;
typedef __attribute__((ext_vector_type(8)))  float  v8f;

#define N_NODES 100000
#define N_EDGES 600000
#define LDIM    128
#define ETILES  (N_EDGES / 16)
#define NTILES  (N_NODES / 16)
#define WAVES   4

__device__ __forceinline__ unsigned short f2bf(float f) {
  unsigned u = __float_as_uint(f);
  u += 0x7FFFu + ((u >> 16) & 1u);          // round-to-nearest-even bf16
  return (unsigned short)(u >> 16);
}

// two f32 -> packed bf16x2 in one dword: vector fptrunc so the backend
// selects a single v_cvt_pk_bf16_f32 (RNE).
__device__ __forceinline__ unsigned pack2(float a, float b) {
  v2f  f = {a, b};
  v2bf r = __builtin_convertvector(f, v2bf);
  return __builtin_bit_cast(unsigned, r);
}

// relu in one VALU op: v_med3_f32(x, 0, +inf)
__device__ __forceinline__ float relu1(float x) {
  return __builtin_amdgcn_fmed3f(x, 0.f, __builtin_inff());
}

// Write a 16x128 f32 result (held as 8 C/D tiles) into LDS in bf16
// A-fragment order for the next layer's K=128 (4 fragments of K=32).
// C/D layout: c[v] = X[v + 8*(lane>>4)][nt*16 + (lane&15)]
// A layout:  element(la, e) with la = M + 16*b, e = 2*(4g+vv)+p,
//            K = 32*kTile + 16g + 8b + 2*vv + p
__device__ __forceinline__ void stash_acc(unsigned short* st, const v8f* acc,
                                          int lane, bool relu) {
  const int n  = lane & 15;
  const int hi = lane >> 4;
  const int p  = n & 1;
  const int b  = (n >> 3) & 1;
  const int vv = (n >> 1) & 3;
#pragma unroll
  for (int nt = 0; nt < 8; ++nt) {
    const int g     = nt & 1;
    const int kTile = nt >> 1;
    const int elem  = ((g * 4 + vv) << 1) | p;
    unsigned short* base = st + kTile * 512 + elem;
#pragma unroll
    for (int v2 = 0; v2 < 4; ++v2) {
      float x0 = acc[nt][2 * v2];
      float x1 = acc[nt][2 * v2 + 1];
      if (relu) { x0 = relu1(x0); x1 = relu1(x1); }
      const unsigned u = pack2(x0, x1);                // one v_cvt_pk_bf16_f32
      const int la0 = (2 * v2 + 8 * hi) + 16 * b;
      base[la0 * 16]       = (unsigned short)u;        // ds_store_b16
      base[(la0 + 1) * 16] = (unsigned short)(u >> 16);// ds_store_b16_d16_hi
    }
  }
}

// One dense layer: D[16x128] = A[16x(32*kcN)] @ W + bias, via bf16 WMMA.
__device__ __forceinline__ void mma_layer(v8f* acc, const unsigned short* st,
                                          const unsigned short* wfr,
                                          const float* __restrict__ bias,
                                          int kcN, int lane) {
  const int n = lane & 15;
#pragma unroll
  for (int nt = 0; nt < 8; ++nt) {
    const float bv = bias[nt * 16 + n];
#pragma unroll
    for (int v = 0; v < 8; ++v) acc[nt][v] = bv;
  }
#pragma unroll 1
  for (int kc = 0; kc < kcN; ++kc) {
    const v16bf a = *reinterpret_cast<const v16bf*>(st + kc * 512 + lane * 16);
#pragma unroll
    for (int nt = 0; nt < 8; ++nt) {
      const v16bf bw = *reinterpret_cast<const v16bf*>(
          wfr + (size_t)(kc * 8 + nt) * 512 + lane * 16);
      acc[nt] = __builtin_amdgcn_wmma_f32_16x16x32_bf16(
          false, a, false, bw, (short)0, acc[nt], false, false);
    }
  }
}

// Pack f32 row-major W[Kdim x 128] into bf16 B-fragment register images:
// dst[frag*512 + lane*16 + e] = W[kc*32 + (e>>3)*16 + (lane>>4)*8 + (e&7)][nt*16 + (lane&15)]
__global__ void pack_weights(const float* __restrict__ W,
                             unsigned short* __restrict__ dst, int Kdim) {
  int id = blockIdx.x * 256 + threadIdx.x;
  if (id >= Kdim * 128) return;
  int frag = id >> 9;
  int rem  = id & 511;
  int la   = rem >> 4;
  int e    = rem & 15;
  int kc   = frag >> 3;
  int nt   = frag & 7;
  int k = kc * 32 + (e >> 3) * 16 + (la >> 4) * 8 + (e & 7);
  int ncol = nt * 16 + (la & 15);
  dst[id] = f2bf(W[(size_t)k * 128 + ncol]);
}

// Fused: gather(sender,edge,receiver) -> edge MLP -> atomic scatter-add.
__global__ __launch_bounds__(WAVES * 32) void edge_mlp_kernel(
    const float* __restrict__ node_lat, const float* __restrict__ edge_lat,
    const int* __restrict__ eidx,
    const unsigned short* __restrict__ w0, const float* __restrict__ b0,
    const unsigned short* __restrict__ w1, const float* __restrict__ b1,
    const unsigned short* __restrict__ w2, const float* __restrict__ b2,
    float* __restrict__ agg) {
  __shared__ __align__(32) unsigned short stage[WAVES][12 * 512];
  __shared__ int meta[WAVES][32];
  const int lane = threadIdx.x & 31;
  const int wid  = threadIdx.x >> 5;
  const int tile = blockIdx.x * WAVES + wid;
  if (tile >= ETILES) return;                       // wave-uniform: EXEC stays all-ones
  unsigned short* st = &stage[wid][0];

  {   // lanes 0..15 load senders, 16..31 receivers for this 16-edge tile
    int m = lane & 15;
    meta[wid][lane] = (lane < 16) ? eidx[tile * 16 + m]
                                  : eidx[N_EDGES + tile * 16 + m];
  }
  asm volatile("s_wait_dscnt 0" ::: "memory");

  // Stage the 16x384 concat input as bf16 A-fragments (12 frags of K=32).
  // 8 consecutive features (aligned) land in 16 contiguous LDS bytes.
#pragma unroll
  for (int j = 0; j < 24; ++j) {
    int gi = lane + 32 * j;          // 0..767 = 16 rows x 48 groups
    int m  = gi / 48;
    int w  = gi - m * 48;
    int k0 = w * 8;
    const float* src;
    if (k0 < 128)      src = node_lat + (size_t)meta[wid][m] * LDIM + k0;
    else if (k0 < 256) src = edge_lat + (size_t)(tile * 16 + m) * LDIM + (k0 - 128);
    else               src = node_lat + (size_t)meta[wid][16 + m] * LDIM + (k0 - 256);
    float4 f0 = *(const float4*)src;
    float4 f1 = *(const float4*)(src + 4);
    uint4 u;
    u.x = pack2(f0.x, f0.y); u.y = pack2(f0.z, f0.w);
    u.z = pack2(f1.x, f1.y); u.w = pack2(f1.z, f1.w);
    int kTile = w >> 2, g = (w >> 1) & 1, b = w & 1;
    int la = m + 16 * b;
    *reinterpret_cast<uint4*>(st + kTile * 512 + la * 16 + g * 8) = u;
  }
  asm volatile("s_wait_dscnt 0" ::: "memory");

  v8f acc[8];
  mma_layer(acc, st, w0, b0, 12, lane);             // 384 -> 128
  stash_acc(st, acc, lane, true);                   // relu
  asm volatile("s_wait_dscnt 0" ::: "memory");
  mma_layer(acc, st, w1, b1, 4, lane);              // 128 -> 128
  stash_acc(st, acc, lane, true);                   // relu
  asm volatile("s_wait_dscnt 0" ::: "memory");
  mma_layer(acc, st, w2, b2, 4, lane);              // 128 -> 128 (no relu)

  // segment_sum via non-returning f32 atomics into agg[receiver][:]
  const int n  = lane & 15;
  const int hi = lane >> 4;
  int recvv[8];
#pragma unroll
  for (int v = 0; v < 8; ++v) recvv[v] = meta[wid][16 + v + 8 * hi];
#pragma unroll
  for (int nt = 0; nt < 8; ++nt) {
#pragma unroll
    for (int v = 0; v < 8; ++v)
      atomicAdd(agg + (size_t)recvv[v] * LDIM + nt * 16 + n, acc[nt][v]);
  }
}

// Fused: concat(node_lat, agg) -> node MLP -> residual -> out (in place over agg).
__global__ __launch_bounds__(WAVES * 32) void node_mlp_kernel(
    const float* __restrict__ node_lat,
    const unsigned short* __restrict__ w0, const float* __restrict__ b0,
    const unsigned short* __restrict__ w1, const float* __restrict__ b1,
    const unsigned short* __restrict__ w2, const float* __restrict__ b2,
    float* __restrict__ out) {
  __shared__ __align__(32) unsigned short stage[WAVES][8 * 512];
  const int lane = threadIdx.x & 31;
  const int wid  = threadIdx.x >> 5;
  const int tile = blockIdx.x * WAVES + wid;
  if (tile >= NTILES) return;
  unsigned short* st = &stage[wid][0];

#pragma unroll
  for (int j = 0; j < 16; ++j) {
    int gi = lane + 32 * j;          // 0..511 = 16 rows x 32 groups
    int m  = gi >> 5;
    int w  = gi & 31;
    int k0 = w * 8;
    const float* src = (k0 < 128)
        ? node_lat + (size_t)(tile * 16 + m) * LDIM + k0
        : out      + (size_t)(tile * 16 + m) * LDIM + (k0 - 128);   // aggregated
    float4 f0 = *(const float4*)src;
    float4 f1 = *(const float4*)(src + 4);
    uint4 u;
    u.x = pack2(f0.x, f0.y); u.y = pack2(f0.z, f0.w);
    u.z = pack2(f1.x, f1.y); u.w = pack2(f1.z, f1.w);
    int kTile = w >> 2, g = (w >> 1) & 1, b = w & 1;
    int la = m + 16 * b;
    *reinterpret_cast<uint4*>(st + kTile * 512 + la * 16 + g * 8) = u;
  }
  asm volatile("s_wait_dscnt 0" ::: "memory");

  v8f acc[8];
  mma_layer(acc, st, w0, b0, 8, lane);              // 256 -> 128
  stash_acc(st, acc, lane, true);
  asm volatile("s_wait_dscnt 0" ::: "memory");
  mma_layer(acc, st, w1, b1, 4, lane);
  stash_acc(st, acc, lane, true);
  asm volatile("s_wait_dscnt 0" ::: "memory");
  mma_layer(acc, st, w2, b2, 4, lane);

  const int n  = lane & 15;
  const int hi = lane >> 4;
#pragma unroll
  for (int nt = 0; nt < 8; ++nt) {
#pragma unroll
    for (int v = 0; v < 8; ++v) {
      size_t o = (size_t)(tile * 16 + v + 8 * hi) * LDIM + nt * 16 + n;
      out[o] = node_lat[o] + acc[nt][v];            // residual, overwrites agg
    }
  }
}

extern "C" void kernel_launch(void* const* d_in, const int* in_sizes, int n_in,
                              void* d_out, int out_size, void* d_ws, size_t ws_size,
                              hipStream_t stream) {
  const float* node_lat = (const float*)d_in[0];
  const float* edge_lat = (const float*)d_in[1];
  const int*   eidx     = (const int*)d_in[2];
  const float* eW0 = (const float*)d_in[4];  const float* eB0 = (const float*)d_in[5];
  const float* eW1 = (const float*)d_in[6];  const float* eB1 = (const float*)d_in[7];
  const float* eW2 = (const float*)d_in[8];  const float* eB2 = (const float*)d_in[9];
  const float* nW0 = (const float*)d_in[10]; const float* nB0 = (const float*)d_in[11];
  const float* nW1 = (const float*)d_in[12]; const float* nB1 = (const float*)d_in[13];
  const float* nW2 = (const float*)d_in[14]; const float* nB2 = (const float*)d_in[15];
  float* out = (float*)d_out;

  // bf16 weight fragments in workspace (147456 ushorts = 288 KB)
  unsigned short* ws   = (unsigned short*)d_ws;
  unsigned short* peW0 = ws;            // 384*128 = 49152
  unsigned short* peW1 = peW0 + 49152;  // 16384
  unsigned short* peW2 = peW1 + 16384;  // 16384
  unsigned short* pnW0 = peW2 + 16384;  // 32768
  unsigned short* pnW1 = pnW0 + 32768;  // 16384
  unsigned short* pnW2 = pnW1 + 16384;  // 16384

  hipMemsetAsync(out, 0, (size_t)N_NODES * LDIM * sizeof(float), stream);
  pack_weights<<<192, 256, 0, stream>>>(eW0, peW0, 384);
  pack_weights<<< 64, 256, 0, stream>>>(eW1, peW1, 128);
  pack_weights<<< 64, 256, 0, stream>>>(eW2, peW2, 128);
  pack_weights<<<128, 256, 0, stream>>>(nW0, pnW0, 256);
  pack_weights<<< 64, 256, 0, stream>>>(nW1, pnW1, 128);
  pack_weights<<< 64, 256, 0, stream>>>(nW2, pnW2, 128);

  edge_mlp_kernel<<<ETILES / WAVES, WAVES * 32, 0, stream>>>(
      node_lat, edge_lat, eidx, peW0, eB0, peW1, eB1, peW2, eB2, out);
  node_mlp_kernel<<<(NTILES + WAVES - 1) / WAVES, WAVES * 32, 0, stream>>>(
      node_lat, pnW0, nB0, pnW1, nB1, pnW2, nB2, out);

  // second output: the ORIGINAL edge latents, passed through
  hipMemcpyAsync(out + (size_t)N_NODES * LDIM, (const void*)edge_lat,
                 (size_t)N_EDGES * LDIM * sizeof(float),
                 hipMemcpyDeviceToDevice, stream);
}
